// HoffmanSwarmV2_4836133175700
// MI455X (gfx1250) — matrix operand order repaired and checked by reading, not compiled, not run on hardware
//
#include <hip/hip_runtime.h>
#include <hip/hip_bf16.h>
#include <math.h>

typedef __bf16 bf16_t;
typedef __attribute__((ext_vector_type(16))) bf16_t v16bf;
typedef __attribute__((ext_vector_type(8)))  bf16_t v8bf;
typedef __attribute__((ext_vector_type(8)))  float  v8f;

#define B_     8
#define N_     1024
#define D_     256
#define P_     64
#define HDC_   4096
#define KMEM_  16
#define BN_    (B_*N_)

// ---------------- WMMA helpers (CDNA5 16x16x32 bf16, fp32 accum) ----------------

__device__ __forceinline__ v8f zero8(){
  v8f z;
#pragma unroll
  for (int i=0;i<8;i++) z[i]=0.f;
  return z;
}

__device__ __forceinline__ v8f wmma_bf(v16bf a, v16bf b, v8f c){
  // (neg_a, A, neg_b, B, c_mod, C, reuse_a, reuse_b)
  return __builtin_amdgcn_wmma_f32_16x16x32_bf16(false, a, false, b, (short)0, c, false, false);
}

// A fragment: 16x32 bf16 tile, row-major source (lda elems). Lane l<16 holds M=l,
// K in {kh*8+0..7} U {16+kh*8+0..7} -> two contiguous 16B loads per lane.
__device__ __forceinline__ v16bf load_a_bf(const bf16_t* A, int lda, int kbase){
  int l = threadIdx.x & 31; int m = l & 15; int kh = l >> 4;
  const bf16_t* p = A + (size_t)m*lda + kbase + kh*8;
  v8bf lo = *(const v8bf*)p;
  v8bf hi = *(const v8bf*)(p + 16);
  v16bf r;
#pragma unroll
  for (int i=0;i<8;i++){ r[i]=lo[i]; r[i+8]=hi[i]; }
  return r;
}

__device__ __forceinline__ v16bf load_a_f32(const float* A, int lda, int kbase){
  int l = threadIdx.x & 31; int m = l & 15; int kh = l >> 4;
  const float* p = A + (size_t)m*lda + kbase + kh*8;
  v16bf r;
#pragma unroll
  for (int i=0;i<8;i++){ r[i]=(bf16_t)p[i]; r[i+8]=(bf16_t)p[i+16]; }
  return r;
}

// B fragment: 32x16 tile from "BT" layout (N-major rows of length ldb along K).
// Lane l<16 = column l, K in {kh*8+0..7} U {16+kh*8+0..7} -> two contiguous 16B loads per lane.
__device__ __forceinline__ v16bf load_b_bf(const bf16_t* BT, int ldb, int kbase){
  int l = threadIdx.x & 31; int n = l & 15; int kh = l >> 4;
  const bf16_t* p = BT + (size_t)n*ldb + kbase + kh*8;
  v8bf lo = *(const v8bf*)p;
  v8bf hi = *(const v8bf*)(p + 16);
  v16bf r;
#pragma unroll
  for (int i=0;i<8;i++){ r[i]=lo[i]; r[i+8]=hi[i]; }
  return r;
}

__device__ __forceinline__ float wave_sum(float v){
#pragma unroll
  for (int off=16; off>0; off>>=1) v += __shfl_xor(v, off);
  return v;
}

// ---------------- K0: weight transpose + f32->bf16 ----------------
__global__ __launch_bounds__(256) void k_wtrans(const float* __restrict__ W,
                                                bf16_t* __restrict__ WT, int K, int N){
  int id = blockIdx.x*256 + threadIdx.x;
  if (id >= K*N) return;
  int k = id / N, n = id % N;
  WT[(size_t)n*K + k] = (bf16_t)W[(size_t)k*N + n];
}

// ---------------- K1: Q/K/C projections (D->P) + C row-normalize ----------------
__global__ __launch_bounds__(128) void k_proj(
    const float* __restrict__ actions,
    const bf16_t* __restrict__ WqT, const float* __restrict__ bq,
    const bf16_t* __restrict__ WkT, const float* __restrict__ bk,
    const bf16_t* __restrict__ WcT, const float* __restrict__ bc,
    bf16_t* __restrict__ Qb, bf16_t* __restrict__ Kb, bf16_t* __restrict__ Cb)
{
  __shared__ float Cs[16][64];
  __shared__ float p2[16][8];
  __shared__ float inv[16];
  int base = blockIdx.x*16;
  int w = threadIdx.x >> 5;
  int l = threadIdx.x & 31; int n_in = l & 15; int kh = l >> 4;
  const float* A = actions + (size_t)base*D_;
  v16bf a[8];
#pragma unroll
  for (int ks=0; ks<8; ks++) a[ks] = load_a_f32(A, D_, ks*32);
  const bf16_t* Ws[3] = {WqT, WkT, WcT};
  const float*  bs[3] = {bq, bk, bc};
  for (int mat=0; mat<3; mat++){
    v8f acc = zero8();
    const bf16_t* BT = Ws[mat] + (size_t)(w*16)*D_;
#pragma unroll
    for (int ks=0; ks<8; ks++) acc = wmma_bf(a[ks], load_b_bf(BT, D_, ks*32), acc);
    int n = w*16 + n_in;
    float bias = bs[mat][n];
#pragma unroll
    for (int r=0; r<8; r++){
      int m = r + 8*kh;
      float v = acc[r] + bias;
      if (mat==0)      Qb[(size_t)(base+m)*P_ + n] = (bf16_t)v;
      else if (mat==1) Kb[(size_t)(base+m)*P_ + n] = (bf16_t)v;
      else             Cs[m][n] = v;
    }
  }
  __syncthreads();
  { int t = threadIdx.x; int r = t >> 3, seg = t & 7;
    float s = 0.f;
#pragma unroll
    for (int i=0;i<8;i++){ float v = Cs[r][seg*8+i]; s += v*v; }
    p2[r][seg] = s; }
  __syncthreads();
  if (threadIdx.x < 16){
    float s=0.f;
#pragma unroll
    for (int i=0;i<8;i++) s += p2[threadIdx.x][i];
    inv[threadIdx.x] = 1.f / fmaxf(sqrtf(s), 1e-12f);
  }
  __syncthreads();
  { int t = threadIdx.x; int r = t >> 3, seg = t & 7;
    float iv = inv[r];
#pragma unroll
    for (int i=0;i<8;i++)
      Cb[(size_t)(base+r)*P_ + seg*8 + i] = (bf16_t)(Cs[r][seg*8+i]*iv); }
}

// ---------------- K2: scores = Q@K^T (WMMA), top-8 + softmax weights ----------------
__global__ __launch_bounds__(256) void k_scores(
    const bf16_t* __restrict__ Qb, const bf16_t* __restrict__ Kb,
    const float* __restrict__ log_temp,
    int* __restrict__ tki, float* __restrict__ tkw)
{
  __shared__ float sc[16*1024];          // 64KB
  int blk = blockIdx.x;
  int b = blk >> 6, rb = blk & 63;
  int base = b*N_ + rb*16;
  int w = threadIdx.x >> 5;
  int l = threadIdx.x & 31; int n_in = l & 15; int kh = l >> 4;
  float tmp = expf(log_temp[0]);
  tmp = fminf(fmaxf(tmp, 0.1f), 10.f);
  float scale = 1.f/(8.f*tmp);           // sqrt(64)=8
  const bf16_t* Arow = Qb + (size_t)base*P_;
  v16bf a0 = load_a_bf(Arow, P_, 0);
  v16bf a1 = load_a_bf(Arow, P_, 32);
  for (int ct = w*8; ct < w*8+8; ct++){
    const bf16_t* BT = Kb + (size_t)(b*N_ + ct*16)*P_;
    v8f acc = zero8();
    acc = wmma_bf(a0, load_b_bf(BT, P_, 0),  acc);
    acc = wmma_bf(a1, load_b_bf(BT, P_, 32), acc);
#pragma unroll
    for (int r=0;r<8;r++) sc[(r+8*kh)*1024 + ct*16 + n_in] = acc[r]*scale;
  }
  __syncthreads();
  // top-8 per row; wave w handles rows w and w+8 (iterative max-extract)
  for (int rl=0; rl<2; rl++){
    int row = w + rl*8;
    float myv = 0.f; int myi = 0;
    for (int it=0; it<8; it++){
      float bv = -3.0e38f; int bi = 0;
      for (int j = l; j < 1024; j += 32){
        float v = sc[row*1024 + j];
        if (v > bv){ bv = v; bi = j; }
      }
#pragma unroll
      for (int off=16; off>0; off>>=1){
        float ov = __shfl_xor(bv, off);
        int   oi = __shfl_xor(bi, off);
        if (ov > bv || (ov == bv && oi < bi)){ bv = ov; bi = oi; }
      }
      if (l == it){ myv = bv; myi = bi; }
      if (l == 0) sc[row*1024 + bi] = -3.0e38f;
      __syncthreads();                    // uniform: 16 barriers for every wave
    }
    float mx = __shfl(myv, 0);            // largest (it==0)
    float e  = (l < 8) ? expf(myv - mx) : 0.f;
    float sum = wave_sum(e);
    if (l < 8){
      tki[(size_t)(base+row)*8 + l] = myi;
      tkw[(size_t)(base+row)*8 + l] = e/sum;
    }
  }
}

// ---------------- K3: messages = (8-sparse adj) @ agent_states; also bf16 transpose ----------------
__global__ __launch_bounds__(256) void k_msg(
    const float* __restrict__ states, const int* __restrict__ tki,
    const float* __restrict__ tkw,
    float* __restrict__ msg, bf16_t* __restrict__ msgT)
{
  __shared__ int   si[128];
  __shared__ float sw[128];
  int blk = blockIdx.x;
  int b = blk >> 6, rb = blk & 63;
  int base = b*N_ + rb*16;
  int t = threadIdx.x;
  if (t < 128){ si[t] = tki[(size_t)base*8 + t]; sw[t] = tkw[(size_t)base*8 + t]; }
  __syncthreads();
  int d = t;
  float acc[16];
#pragma unroll
  for (int r=0;r<16;r++){
    float s = 0.f;
#pragma unroll
    for (int i=0;i<8;i++)
      s += sw[r*8+i]*states[((size_t)b*N_ + si[r*8+i])*D_ + d];
    acc[r] = s;
    msg[(size_t)(base+r)*D_ + d] = s;
  }
  v16bf tb;
#pragma unroll
  for (int r=0;r<16;r++) tb[r] = (bf16_t)acc[r];
  *(v16bf*)(msgT + ((size_t)b*D_ + d)*N_ + rb*16) = tb;   // [b][d][n], 32B aligned
}

// ---------------- K4: sim (WMMA) -> sigmoid coalition -> cw@messages (WMMA) -> mix ----------------
__global__ __launch_bounds__(256) void k_mix(
    const bf16_t* __restrict__ Cb, const float* __restrict__ ctemp,
    const float* __restrict__ msg, const bf16_t* __restrict__ msgT,
    float* __restrict__ sp, bf16_t* __restrict__ spb)
{
  __shared__ bf16_t cw[16*1024];          // 32KB
  __shared__ float rsum[16];
  int blk = blockIdx.x;
  int b = blk >> 6, rb = blk & 63;
  int base = b*N_ + rb*16;
  int w = threadIdx.x >> 5;
  int l = threadIdx.x & 31; int n_in = l & 15; int kh = l >> 4;
  float ctv = ctemp[0];
  const bf16_t* Arow = Cb + (size_t)base*P_;
  v16bf a0 = load_a_bf(Arow, P_, 0);
  v16bf a1 = load_a_bf(Arow, P_, 32);
  for (int ct = w*8; ct < w*8+8; ct++){
    const bf16_t* BT = Cb + (size_t)(b*N_ + ct*16)*P_;
    v8f acc = zero8();
    acc = wmma_bf(a0, load_b_bf(BT, P_, 0),  acc);
    acc = wmma_bf(a1, load_b_bf(BT, P_, 32), acc);
#pragma unroll
    for (int r=0;r<8;r++){
      float v = 1.f/(1.f + expf(-(acc[r]-0.7f)*ctv));
      cw[(r+8*kh)*1024 + ct*16 + n_in] = (bf16_t)v;
    }
  }
  __syncthreads();
  for (int rl=0; rl<2; rl++){
    int row = w + rl*8;
    float s = 0.f;
    for (int j=l; j<1024; j+=32) s += (float)cw[row*1024+j];
    s = wave_sum(s);
    if (l==0) rsum[row] = s;
  }
  __syncthreads();
  for (int nt = w*2; nt < w*2+2; nt++){
    v8f acc = zero8();
    const bf16_t* BT = msgT + (size_t)b*D_*N_ + (size_t)(nt*16)*N_;
    for (int ks=0; ks<32; ks++){
      v16bf a = load_a_bf((const bf16_t*)cw, 1024, ks*32);   // A fragments from LDS
      acc = wmma_bf(a, load_b_bf(BT, N_, ks*32), acc);
    }
    int n = nt*16 + n_in;
#pragma unroll
    for (int r=0;r<8;r++){
      int m = r + 8*kh;
      float val = 0.8f*msg[(size_t)(base+m)*D_ + n] + 0.2f*acc[r]/(rsum[m]+1e-8f);
      sp[(size_t)(base+m)*D_ + n]  = val;
      spb[(size_t)(base+m)*D_ + n] = (bf16_t)val;
    }
  }
}

// ---------------- K5: generic (8192x256)@(256x256)+bias GEMM ----------------
__global__ __launch_bounds__(256) void k_gemm256(
    const bf16_t* __restrict__ A, const bf16_t* __restrict__ BT,
    const float* __restrict__ bias, float* __restrict__ out)
{
  int base = blockIdx.x*16;
  int w = threadIdx.x >> 5;
  int l = threadIdx.x & 31; int n_in = l & 15; int kh = l >> 4;
  const bf16_t* Arow = A + (size_t)base*D_;
  v16bf a[8];
#pragma unroll
  for (int ks=0;ks<8;ks++) a[ks] = load_a_bf(Arow, D_, ks*32);
  for (int nt = w*2; nt < w*2+2; nt++){
    v8f acc = zero8();
    const bf16_t* Bp = BT + (size_t)(nt*16)*D_;
#pragma unroll
    for (int ks=0;ks<8;ks++) acc = wmma_bf(a[ks], load_b_bf(Bp, D_, ks*32), acc);
    int n = nt*16 + n_in;
    float bb = bias[n];
#pragma unroll
    for (int r=0;r<8;r++)
      out[(size_t)(base + r + 8*kh)*D_ + n] = acc[r] + bb;
  }
}

// ---------------- K6: per-row fused k/v projection (WMMA) + 16-key MHA ----------------
__global__ __launch_bounds__(256) void k_attn(
    const float* __restrict__ membuf,
    const bf16_t* __restrict__ WakT, const float* __restrict__ bak,
    const bf16_t* __restrict__ WavT, const float* __restrict__ bav,
    const float* __restrict__ q2,
    float* __restrict__ oat, bf16_t* __restrict__ oab)
{
  __shared__ float K16[16*256];
  __shared__ float V16[16*256];
  __shared__ float sb[64];
  __shared__ float aw[64];
  int bn = blockIdx.x;
  int w = threadIdx.x >> 5;
  int l = threadIdx.x & 31; int n_in = l & 15; int kh = l >> 4;
  const float* A = membuf + (size_t)bn*KMEM_*D_;   // 16x256 tile = the M-block
  v16bf a[8];
#pragma unroll
  for (int ks=0;ks<8;ks++) a[ks] = load_a_f32(A, D_, ks*32);
  for (int t4=0; t4<4; t4++){
    int tid = w*4 + t4;                  // 32 tiles: 16 for K-proj, 16 for V-proj
    int mat = tid >> 4, nt = tid & 15;
    const bf16_t* BT = (mat ? WavT : WakT) + (size_t)(nt*16)*D_;
    v8f acc = zero8();
#pragma unroll
    for (int ks=0;ks<8;ks++) acc = wmma_bf(a[ks], load_b_bf(BT, D_, ks*32), acc);
    int n = nt*16 + n_in;
    float bb = (mat ? bav : bak)[n];
    float* dst = mat ? V16 : K16;
#pragma unroll
    for (int r=0;r<8;r++) dst[(r+8*kh)*256 + n] = acc[r] + bb;
  }
  __syncthreads();
  int t = threadIdx.x;
  if (t < 64){
    int h = t >> 4, j = t & 15;
    const float* q  = q2 + (size_t)bn*D_ + h*64;
    const float* kr = K16 + j*256 + h*64;
    float s = 0.f;
#pragma unroll
    for (int d=0; d<64; d++) s += q[d]*kr[d];
    sb[t] = s*0.125f;                    // 1/sqrt(64)
  }
  __syncthreads();
  if (t < 4){
    float mx = -3e38f;
    for (int j=0;j<16;j++) mx = fmaxf(mx, sb[t*16+j]);
    float e[16]; float sum = 0.f;
    for (int j=0;j<16;j++){ e[j]=expf(sb[t*16+j]-mx); sum += e[j]; }
    for (int j=0;j<16;j++) aw[t*16+j] = e[j]/sum;
  }
  __syncthreads();
  { int h = t >> 6;
    float s = 0.f;
#pragma unroll
    for (int j=0;j<16;j++) s += aw[h*16+j]*V16[j*256 + t];
    oat[(size_t)bn*D_ + t] = s;
    oab[(size_t)bn*D_ + t] = (bf16_t)s; }
}

// ---------------- K7: residual + LayerNorm ----------------
__global__ __launch_bounds__(256) void k_ln(
    const float* __restrict__ sp, const float* __restrict__ xo,
    const float* __restrict__ g, const float* __restrict__ bp,
    float* __restrict__ att, bf16_t* __restrict__ attb)
{
  __shared__ float red[8];
  int bn = blockIdx.x;
  int t = threadIdx.x;
  int w = t >> 5, l = t & 31;
  float x = sp[(size_t)bn*D_ + t] + xo[(size_t)bn*D_ + t];
  float s = wave_sum(x);
  if (l==0) red[w] = s;
  __syncthreads();
  float tot = 0.f;
#pragma unroll
  for (int i=0;i<8;i++) tot += red[i];
  float mean = tot*(1.f/256.f);
  __syncthreads();
  float d = x - mean;
  float s2 = wave_sum(d*d);
  if (l==0) red[w] = s2;
  __syncthreads();
  float tot2 = 0.f;
#pragma unroll
  for (int i=0;i<8;i++) tot2 += red[i];
  float var = tot2*(1.f/256.f);
  float y = d * rsqrtf(var + 1e-5f) * g[t] + bp[t];
  att[(size_t)bn*D_ + t]  = y;
  attb[(size_t)bn*D_ + t] = (bf16_t)y;
}

// ---------------- K8: fused Whdc GEMM (WMMA) + STE sign + gate + gated write + cosine read ----------------
__global__ __launch_bounds__(256) void k_hdc(
    const bf16_t* __restrict__ attb, const bf16_t* __restrict__ WhdcT,
    const float* __restrict__ bhdc, const float* __restrict__ att,
    const float* __restrict__ hdcmem, const float* __restrict__ pcodes,
    const float* __restrict__ Wg, const float* __restrict__ bg,
    const int* __restrict__ step,
    float* __restrict__ newmem, float* __restrict__ rsim)
{
  __shared__ float gpart[256];
  __shared__ float gate[16];
  __shared__ float dota[16];
  __shared__ float nsqa[16];
  int base = blockIdx.x*16;
  int t = threadIdx.x;
  int w = t >> 5;
  int l = t & 31; int n_in = l & 15; int kh = l >> 4;
  if (t < 16){ dota[t]=0.f; nsqa[t]=0.f; }
  // gate = sigmoid([attended, hdc_memory] @ Wg + bg) — independent of hf
  { int r = t >> 4, p = t & 15;
    const float* arow = att    + (size_t)(base+r)*D_;
    const float* hrow = hdcmem + (size_t)(base+r)*HDC_;
    float s = 0.f;
#pragma unroll
    for (int i=0;i<16;i++) s += arow[p*16+i]*Wg[p*16+i];
    for (int i=0;i<256;i++) s += hrow[p*256+i]*Wg[256 + p*256+i];
    gpart[t] = s; }
  __syncthreads();
  if (t < 16){
    float s = 0.f;
#pragma unroll
    for (int p=0;p<16;p++) s += gpart[t*16+p];
    gate[t] = 1.f/(1.f + expf(-(s + bg[0])));
  }
  __syncthreads();
  int sm = ((step[0] % 256) + 256) % 256;
  const float* pos = pcodes + (size_t)sm*HDC_;
  const bf16_t* Arow = attb + (size_t)base*D_;
  v16bf a[8];
#pragma unroll
  for (int ks=0;ks<8;ks++) a[ks] = load_a_bf(Arow, D_, ks*32);
  float dotp[8], nsqp[8];
#pragma unroll
  for (int r=0;r<8;r++){ dotp[r]=0.f; nsqp[r]=0.f; }
  for (int t8=0; t8<32; t8++){
    int nt = w*32 + t8;                         // 256 N-tiles over HDC=4096
    const bf16_t* BT = WhdcT + (size_t)(nt*16)*D_;
    v8f acc = zero8();
#pragma unroll
    for (int ks=0;ks<8;ks++) acc = wmma_bf(a[ks], load_b_bf(BT, D_, ks*32), acc);
    int n = nt*16 + n_in;
    float bb = bhdc[n];
    float pc = pos[n];
#pragma unroll
    for (int r=0;r<8;r++){
      int m = r + 8*kh;
      size_t idx = (size_t)(base+m)*HDC_ + n;
      float hf = acc[r] + bb;
      float hv = (hf >= 0.f) ? 1.f : -1.f;      // STE forward == sign (0 -> +1)
      float bd = hv * pc;
      float hm = hdcmem[idx];
      float gv = gate[m];
      float nm = gv*hm + (1.f-gv)*bd;
      newmem[idx] = nm;
      dotp[r] += nm*hv;
      nsqp[r] += nm*nm;
    }
  }
#pragma unroll
  for (int r=0;r<8;r++){
    atomicAdd(&dota[r+8*kh], dotp[r]);
    atomicAdd(&nsqa[r+8*kh], nsqp[r]);
  }
  __syncthreads();
  if (t < 16)
    rsim[base+t] = dota[t] / fmaxf(sqrtf(nsqa[t])*64.0f, 1e-8f);  // ||hdc_vec||=sqrt(4096)
}

// ---------------- launch ----------------
extern "C" void kernel_launch(void* const* d_in, const int* in_sizes, int n_in,
                              void* d_out, int out_size, void* d_ws, size_t ws_size,
                              hipStream_t stream)
{
  (void)in_sizes; (void)n_in; (void)out_size; (void)ws_size;
  const float* agent_states  = (const float*)d_in[0];
  const float* agent_actions = (const float*)d_in[1];
  const float* membuf        = (const float*)d_in[2];
  const float* pcodes        = (const float*)d_in[3];
  const float* hdcmem        = (const float*)d_in[4];
  const float* Wq  = (const float*)d_in[5];  const float* bq  = (const float*)d_in[6];
  const float* Wk  = (const float*)d_in[7];  const float* bk  = (const float*)d_in[8];
  const float* logt= (const float*)d_in[9];
  const float* Wc  = (const float*)d_in[10]; const float* bc  = (const float*)d_in[11];
  const float* ctemp=(const float*)d_in[12];
  const float* Waq = (const float*)d_in[13]; const float* baq = (const float*)d_in[14];
  const float* Wak = (const float*)d_in[15]; const float* bak = (const float*)d_in[16];
  const float* Wav = (const float*)d_in[17]; const float* bav = (const float*)d_in[18];
  const float* Wao = (const float*)d_in[19]; const float* bao = (const float*)d_in[20];
  const float* lng = (const float*)d_in[21]; const float* lnb = (const float*)d_in[22];
  const float* Whdc= (const float*)d_in[23]; const float* bhdc= (const float*)d_in[24];
  const float* Wgp = (const float*)d_in[25]; const float* bg  = (const float*)d_in[26];
  const int*   step= (const int*)d_in[27];

  char* ws = (char*)d_ws;
  size_t off = 0;
  auto alloc = [&](size_t bytes)->char* {
    char* p = ws + off; off += (bytes + 255) & ~(size_t)255; return p;
  };
  bf16_t* WqT  = (bf16_t*)alloc((size_t)P_*D_*2);
  bf16_t* WkT  = (bf16_t*)alloc((size_t)P_*D_*2);
  bf16_t* WcT  = (bf16_t*)alloc((size_t)P_*D_*2);
  bf16_t* WaqT = (bf16_t*)alloc((size_t)D_*D_*2);
  bf16_t* WakT = (bf16_t*)alloc((size_t)D_*D_*2);
  bf16_t* WavT = (bf16_t*)alloc((size_t)D_*D_*2);
  bf16_t* WaoT = (bf16_t*)alloc((size_t)D_*D_*2);
  bf16_t* WhdcT= (bf16_t*)alloc((size_t)HDC_*D_*2);
  bf16_t* Qb   = (bf16_t*)alloc((size_t)BN_*P_*2);
  bf16_t* Kb   = (bf16_t*)alloc((size_t)BN_*P_*2);
  bf16_t* Cb   = (bf16_t*)alloc((size_t)BN_*P_*2);
  int*    tki  = (int*)   alloc((size_t)BN_*8*4);
  float*  tkw  = (float*) alloc((size_t)BN_*8*4);
  float*  msg  = (float*) alloc((size_t)BN_*D_*4);
  bf16_t* msgT = (bf16_t*)alloc((size_t)B_*D_*N_*2);
  float*  sp   = (float*) alloc((size_t)BN_*D_*4);
  bf16_t* spb  = (bf16_t*)alloc((size_t)BN_*D_*2);
  float*  q2   = (float*) alloc((size_t)BN_*D_*4);
  float*  oat  = (float*) alloc((size_t)BN_*D_*4);
  bf16_t* oab  = (bf16_t*)alloc((size_t)BN_*D_*2);
  float*  xo   = (float*) alloc((size_t)BN_*D_*4);
  bf16_t* attb = (bf16_t*)alloc((size_t)BN_*D_*2);

  float* out_att = (float*)d_out;                       // (8,1024,256)
  float* out_nm  = out_att + (size_t)BN_*D_;            // (8192,4096)
  float* out_rs  = out_nm  + (size_t)BN_*HDC_;          // (8192,)

  k_wtrans<<<64,  256, 0, stream>>>(Wq,  WqT,  D_, P_);
  k_wtrans<<<64,  256, 0, stream>>>(Wk,  WkT,  D_, P_);
  k_wtrans<<<64,  256, 0, stream>>>(Wc,  WcT,  D_, P_);
  k_wtrans<<<256, 256, 0, stream>>>(Waq, WaqT, D_, D_);
  k_wtrans<<<256, 256, 0, stream>>>(Wak, WakT, D_, D_);
  k_wtrans<<<256, 256, 0, stream>>>(Wav, WavT, D_, D_);
  k_wtrans<<<256, 256, 0, stream>>>(Wao, WaoT, D_, D_);
  k_wtrans<<<4096,256, 0, stream>>>(Whdc,WhdcT,D_, HDC_);

  k_proj   <<<512, 128, 0, stream>>>(agent_actions, WqT,bq, WkT,bk, WcT,bc, Qb,Kb,Cb);
  k_scores <<<512, 256, 0, stream>>>(Qb, Kb, logt, tki, tkw);
  k_msg    <<<512, 256, 0, stream>>>(agent_states, tki, tkw, msg, msgT);
  k_mix    <<<512, 256, 0, stream>>>(Cb, ctemp, msg, msgT, sp, spb);
  k_gemm256<<<512, 256, 0, stream>>>(spb, WaqT, baq, q2);
  k_attn   <<<8192,256, 0, stream>>>(membuf, WakT,bak, WavT,bav, q2, oat, oab);
  k_gemm256<<<512, 256, 0, stream>>>(oab, WaoT, bao, xo);
  k_ln     <<<8192,256, 0, stream>>>(sp, xo, lng, lnb, out_att, attb);
  k_hdc    <<<512, 256, 0, stream>>>(attb, WhdcT, bhdc, out_att, hdcmem, pcodes,
                                     Wgp, bg, step, out_nm, out_rs);
}
